// SpatialAttention_68152541053305
// MI455X (gfx1250) — matrix operand order repaired and checked by reading, not compiled, not run on hardware
//
#include <hip/hip_runtime.h>
#include <hip/hip_bf16.h>
#include <math.h>

typedef __attribute__((ext_vector_type(16))) __bf16 v16bf;
typedef __attribute__((ext_vector_type(8)))  __bf16 v8bf;
typedef __attribute__((ext_vector_type(4)))  __bf16 v4bf;
typedef __attribute__((ext_vector_type(8)))  float  v8f;
typedef __attribute__((ext_vector_type(4)))  float  v4f;

#define B_SZ 8
#define L_SZ 1024
#define NH   16
#define NG   4
#define HD   64
#define QKVO 1536
#define DMODEL 1024

// ---------------- WMMA helpers (gfx1250 wave32, ISA 7.12.2 layouts) ----------------

__device__ __forceinline__ v8f wmma_bf16(v16bf a, v16bf b, v8f c) {
  return __builtin_amdgcn_wmma_f32_16x16x32_bf16(false, a, false, b, (short)0, c,
                                                 false, false);
}

// A-matrix 16x32 bf16 fragment: lane holds row M=lane&15.
// K = half*16 + {0..15} (matches B fragment order).
__device__ __forceinline__ v16bf load_A_frag(const __bf16* rowp, int hlf) {
  v8bf c0 = *(const v8bf*)(rowp + hlf * 16);
  v8bf c1 = *(const v8bf*)(rowp + hlf * 16 + 8);
  return __builtin_shufflevector(c0, c1, 0, 1, 2, 3, 4, 5, 6, 7,
                                 8, 9, 10, 11, 12, 13, 14, 15);
}

// B-matrix 32x16 bf16 fragment: lane holds col N=lane&15, K = half*16 + {0..15}.
__device__ __forceinline__ v16bf load_B_frag(const __bf16* rowp, int hlf) {
  v8bf c0 = *(const v8bf*)(rowp + hlf * 16);
  v8bf c1 = *(const v8bf*)(rowp + hlf * 16 + 8);
  return __builtin_shufflevector(c0, c1, 0, 1, 2, 3, 4, 5, 6, 7,
                                 8, 9, 10, 11, 12, 13, 14, 15);
}

// ---------------- Kernel 1/4: C[M,N] = A[M,K] * W[N,K]^T  (f32 in, f32 out) -------
// N, K compile-time so row*N / row*K fold into vmem immediate offsets.

#define BM 128
#define BN 64
#define BK 32
#define LDT 40  // BK + 8 pad: row stride 80B -> every 16-elem chunk 16B aligned

template <int N, int K>
__global__ __launch_bounds__(256) void gemm_bf16_wmma(
    const float* __restrict__ A, const float* __restrict__ W,
    float* __restrict__ C) {
  __shared__ __align__(16) __bf16 As[BM][LDT];
  __shared__ __align__(16) __bf16 Bs[BN][LDT];

  const int tid  = threadIdx.x;
  const int lane = tid & 31;
  const int wid  = tid >> 5;
  const int hlf  = lane >> 4;
  const int l16  = lane & 15;
  const int bm   = blockIdx.y * BM;
  const int bn   = blockIdx.x * BN;
  const int wm   = (wid >> 1) * 32;  // 0..96
  const int wn   = (wid & 1) * 32;   // 0,32

  // per-thread staging coords (constant across K loop)
  const int ar = tid >> 3, ac = (tid & 7) * 4;   // A: rows ar, ar+32, ar+64, ar+96
  const float* Arow = A + (size_t)(bm + ar) * K + ac;
  const float* Wrow = W + (size_t)(bn + ar) * K + ac;  // B: rows ar, ar+32

  v8f acc[2][2] = {};

  for (int k0 = 0; k0 < K; k0 += BK) {
    // stage via registers: 4+2 vector b128 loads pipeline under one waitcnt,
    // then packed bf16 ds_store_b64 (v_cvt_pk_bf16_f32 pairs).
    v4f areg[4], breg[2];
#pragma unroll
    for (int i = 0; i < 4; ++i)
      areg[i] = *(const v4f*)(Arow + (size_t)i * 32 * K + k0);
#pragma unroll
    for (int i = 0; i < 2; ++i)
      breg[i] = *(const v4f*)(Wrow + (size_t)i * 32 * K + k0);
    if (k0 + BK < K) {  // global_prefetch_b8 of next tiles
      __builtin_prefetch(Arow + k0 + BK, 0, 0);
      __builtin_prefetch(Wrow + k0 + BK, 0, 0);
    }
#pragma unroll
    for (int i = 0; i < 4; ++i)
      *(v4bf*)&As[ar + i * 32][ac] = __builtin_convertvector(areg[i], v4bf);
#pragma unroll
    for (int i = 0; i < 2; ++i)
      *(v4bf*)&Bs[ar + i * 32][ac] = __builtin_convertvector(breg[i], v4bf);
    __syncthreads();

    v16bf af[2], bfr[2];
#pragma unroll
    for (int t = 0; t < 2; ++t) {
      af[t]  = load_A_frag(&As[wm + t * 16 + l16][0], hlf);
      bfr[t] = load_B_frag(&Bs[wn + t * 16 + l16][0], hlf);
    }
#pragma unroll
    for (int i = 0; i < 2; ++i)
#pragma unroll
      for (int j = 0; j < 2; ++j)
        acc[i][j] = wmma_bf16(af[i], bfr[j], acc[i][j]);
    __syncthreads();
  }

  // C layout: lane holds col N=lane&15, rows M = 8*half + r.
  // One base pointer; all row/tile displacements are compile-time immediates.
  float* cbase = C + (size_t)(bm + wm + hlf * 8) * N + bn + wn + l16;
#pragma unroll
  for (int i = 0; i < 2; ++i)
#pragma unroll
    for (int j = 0; j < 2; ++j)
#pragma unroll
      for (int r = 0; r < 8; ++r)
        cbase[(size_t)(i * 16 + r) * N + j * 16] = acc[i][j][r];
}

// ---------------- Kernel 2: qk-norm + RoPE(p=h+w) + pack bf16, V transposed -------

__global__ __launch_bounds__(256) void normrope_pack(
    const float* __restrict__ qkv, __bf16* __restrict__ Qb,
    __bf16* __restrict__ Kb, __bf16* __restrict__ Vt) {
  const int warp = (int)((blockIdx.x * blockDim.x + threadIdx.x) >> 5);
  const int lane = threadIdx.x & 31;
  if (warp >= B_SZ * L_SZ * 24) return;
  const int slot = warp % 24;            // 0..15 q-head, 16..19 k-group, 20..23 v-group
  const int l    = (warp / 24) % L_SZ;
  const int b    = warp / (24 * L_SZ);

  const size_t base = (size_t)(b * L_SZ + l) * QKVO;
  const float* src;
  if (slot < 16)      src = qkv + base + slot * HD;
  else if (slot < 20) src = qkv + base + 1024 + (slot - 16) * HD;
  else                src = qkv + base + 1280 + (slot - 20) * HD;

  float x1 = src[2 * lane];
  float x2 = src[2 * lane + 1];

  if (slot < 20) {
    float ss = x1 * x1 + x2 * x2;
#pragma unroll
    for (int m = 16; m >= 1; m >>= 1) ss += __shfl_xor(ss, m, 32);
    float inv = 1.0f / (sqrtf(ss) + 1e-10f);
    x1 *= inv; x2 *= inv;
    // RoPE: freqs = (h + w) * inv_freq, inv_freq[f] = 10000^(-2f/64)
    float p   = (float)(l >> 5) + (float)(l & 31);
    float fr  = __expf(-((float)(2 * lane) / 64.0f) * 9.210340371976184f);  // ln(1e4)
    float ang = p * fr;
    float c = __cosf(ang), s = __sinf(ang);
    float o1 = x1 * c - x2 * s;   // -> dim f
    float o2 = x1 * s + x2 * c;   // -> dim 32+f
    if (slot < 16) {
      o1 *= 0.125f; o2 *= 0.125f;  // fold 1/sqrt(64) into Q
      __bf16* dst = Qb + ((size_t)(b * NH + slot) * L_SZ + l) * HD;
      dst[lane] = (__bf16)o1; dst[32 + lane] = (__bf16)o2;
    } else {
      __bf16* dst = Kb + ((size_t)(b * NG + (slot - 16)) * L_SZ + l) * HD;
      dst[lane] = (__bf16)o1; dst[32 + lane] = (__bf16)o2;
    }
  } else {
    // V transposed: Vt[b][g][d][l]
    __bf16* dst = Vt + (size_t)(b * NG + (slot - 20)) * HD * L_SZ;
    dst[(size_t)(2 * lane) * L_SZ + l]     = (__bf16)x1;
    dst[(size_t)(2 * lane + 1) * L_SZ + l] = (__bf16)x2;
  }
}

// ---------------- Kernel 3: flash attention, fully transposed, no LDS ------------
// S^T = K Q^T  (lane owns one query column); O^T = V^T P^T.
// 128-thread block = 4 waves = 1 wave per SIMD of a WGP; full VGPR budget.

__global__ __launch_bounds__(128, 1) void attn_fa(
    const __bf16* __restrict__ Qb, const __bf16* __restrict__ Kb,
    const __bf16* __restrict__ Vt, float* __restrict__ Out) {
  const int tid = threadIdx.x, lane = tid & 31, wid = tid >> 5;
  const int hlf = lane >> 4, l16 = lane & 15;
  const int bh = blockIdx.y;  // b*16 + h
  const int b = bh >> 4, h = bh & 15, g = h >> 2;
  const int q0 = blockIdx.x * 64 + wid * 16;

  const __bf16* qbase = Qb + ((size_t)bh * L_SZ + q0) * HD;
  const __bf16* kbase = Kb + (size_t)(b * NG + g) * L_SZ * HD;
  const __bf16* vbase = Vt + (size_t)(b * NG + g) * HD * L_SZ;

  // Q^T as B fragments: lane = query col l16, dims 16*hlf..+15 per 32-dim chunk
  v16bf qb[2];
#pragma unroll
  for (int kf = 0; kf < 2; ++kf)
    qb[kf] = load_B_frag(qbase + (size_t)l16 * HD + kf * 32, hlf);

  v8f o[4] = {};                 // O^T: lane = query, rows = dims 16t + 8*hlf + r
  float mrow = -INFINITY, lrow = 0.0f;

  for (int kv = 0; kv < L_SZ; kv += 32) {
    // S^T tiles: s[c] rows = keys kv + 16c + 8*hlf + r, col = query l16
    v8f s[2] = {};
#pragma unroll
    for (int c = 0; c < 2; ++c)
#pragma unroll
      for (int kf = 0; kf < 2; ++kf) {
        v16bf ka = load_A_frag(kbase + (size_t)(kv + c * 16 + l16) * HD + kf * 32, hlf);
        s[c] = wmma_bf16(ka, qb[kf], s[c]);
      }

    // online softmax over this 32-key slab (per query = per lane column)
    float mx = s[0][0];
#pragma unroll
    for (int c = 0; c < 2; ++c)
#pragma unroll
      for (int r = 0; r < 8; ++r) mx = fmaxf(mx, s[c][r]);
    mx = fmaxf(mx, __shfl_xor(mx, 16, 32));      // other 16 keys of the slab
    float mnew  = fmaxf(mrow, mx);
    float alpha = __expf(mrow - mnew);
    mrow = mnew;

    float p0[8], p1[8], rs = 0.0f;
#pragma unroll
    for (int r = 0; r < 8; ++r) {
      p0[r] = __expf(s[0][r] - mnew);
      p1[r] = __expf(s[1][r] - mnew);
      rs += p0[r] + p1[r];
    }
    rs += __shfl_xor(rs, 16, 32);
    lrow = lrow * alpha + rs;
#pragma unroll
    for (int t = 0; t < 4; ++t)
#pragma unroll
      for (int r = 0; r < 8; ++r) o[t][r] *= alpha;

    // Assemble P^T B-fragment: lane needs keys 16*hlf + {0..15} of its query.
    // Select-before-shuffle (8 shuffles), and crucially all vector subscripts
    // are compile-time constants: the hlf selection is applied to the VALUES
    // (v_cndmask), never to the vector index, so pb stays in VGPRs.
    v16bf pb;
#pragma unroll
    for (int r = 0; r < 8; ++r) {
      float own = hlf ? p1[r] : p0[r];           // my tile c = hlf, my half
      float snd = hlf ? p0[r] : p1[r];           // what my partner needs
      float oth = __shfl_xor(snd, 16, 32);       // partner's p[my hlf][r]
      pb[r]     = (__bf16)(hlf ? oth : own);     // keys 16*hlf + 0..7
      pb[8 + r] = (__bf16)(hlf ? own : oth);     // keys 16*hlf + 8..15
    }

    // O^T += V^T P^T : A = V^T (dims x 32 keys, contiguous), shared B = P^T
#pragma unroll
    for (int t = 0; t < 4; ++t) {
      v16bf va = load_A_frag(vbase + (size_t)(t * 16 + l16) * L_SZ + kv, hlf);
      o[t] = wmma_bf16(va, pb, o[t]);
    }
  }

  // epilogue: lane owns query row q0+l16 -> contiguous float4 stores
  float inv = 1.0f / lrow;
  float* obase = Out + (size_t)(b * L_SZ + q0 + l16) * DMODEL + h * HD;
#pragma unroll
  for (int t = 0; t < 4; ++t) {
    v4f lo, hi;
#pragma unroll
    for (int j = 0; j < 4; ++j) { lo[j] = o[t][j] * inv; hi[j] = o[t][4 + j] * inv; }
    *(v4f*)(obase + t * 16 + hlf * 8)     = lo;
    *(v4f*)(obase + t * 16 + hlf * 8 + 4) = hi;
  }
}

// ---------------- launch ----------------

extern "C" void kernel_launch(void* const* d_in, const int* in_sizes, int n_in,
                              void* d_out, int out_size, void* d_ws, size_t ws_size,
                              hipStream_t stream) {
  const float* x     = (const float*)d_in[0];
  const float* w_qkv = (const float*)d_in[1];
  const float* w_o   = (const float*)d_in[2];
  // H_patch = W_patch = 32 hardcoded per reference constants.

  char* p = (char*)d_ws;
  float* qkv = (float*)p;   p += (size_t)B_SZ * L_SZ * QKVO * sizeof(float);   // 48MB
  __bf16* Qb = (__bf16*)p;  p += (size_t)B_SZ * NH * L_SZ * HD * 2;            // 16MB
  __bf16* Kb = (__bf16*)p;  p += (size_t)B_SZ * NG * L_SZ * HD * 2;            //  4MB
  __bf16* Vt = (__bf16*)p;  p += (size_t)B_SZ * NG * HD * L_SZ * 2;            //  4MB
  float* attn = (float*)p;                                                     // 32MB

  const int M = B_SZ * L_SZ;  // 8192
  gemm_bf16_wmma<QKVO, DMODEL><<<dim3(QKVO / BN, M / BM), 256, 0, stream>>>(
      x, w_qkv, qkv);
  normrope_pack<<<dim3((B_SZ * L_SZ * 24) / 8), 256, 0, stream>>>(qkv, Qb, Kb, Vt);
  attn_fa<<<dim3(L_SZ / 64, B_SZ * NH), 128, 0, stream>>>(Qb, Kb, Vt, attn);
  gemm_bf16_wmma<DMODEL, DMODEL><<<dim3(DMODEL / BN, M / BM), 256, 0, stream>>>(
      attn, w_o, (float*)d_out);
}
